// RNNBase_26542897889417
// MI455X (gfx1250) — matrix operand order repaired and checked by reading, not compiled, not run on hardware
//
#include <hip/hip_runtime.h>
#include <hip/hip_bf16.h>

// Problem dims
#define BB 64
#define TT 512
#define DD 256
#define HH 1024
#define G4 4096          // 4*H
#define KC 1280          // D + H (concatenated K of the fused step GEMM)
#define NKB (KC / 32)    // 40 K-blocks per step
#define BTILE (4 * 16 * 32)  // one staged B tile set: 4 gates x 16 cols x 32 K

typedef __bf16 bf16;
typedef __attribute__((ext_vector_type(16))) __bf16 v16bf;
typedef __attribute__((ext_vector_type(8)))  __bf16 v8bf;
typedef __attribute__((ext_vector_type(8)))  float  v8f;

static __device__ __forceinline__ v8bf ld8g(const bf16* p) { return *(const v8bf*)p; }

static __device__ __forceinline__ v16bf cat16(v8bf lo, v8bf hi) {
  return __builtin_shufflevector(lo, hi, 0,1,2,3,4,5,6,7,8,9,10,11,12,13,14,15);
}

// A fragment (16x32 bf16, row-major source, per-lane row already folded into rowptr).
// A k order used here, matched by B: lanes 0-15 hold K 0..15, lanes 16-31 hold K 16..31 (contiguous).
static __device__ __forceinline__ v16bf load_a_frag(const bf16* rowptr, int lane) {
  const int kk = (lane < 16) ? 0 : 16;
  return cat16(ld8g(rowptr + kk), ld8g(rowptr + kk + 8));
}

static __device__ __forceinline__ float sigmoidf_(float x) {
  return 1.0f / (1.0f + __expf(-x));
}

// ---------------------------------------------------------------------------
// Per-timestep fused LSTM step:
//   gates = [x_t , h_{t-1}] * Wcat^T + (bx+bh);  all-sigmoid gates (source quirk)
//   c = c*f + i*g;  h = o*tanh(c)
// One block per 16-wide hidden tile; wave w handles batch rows [16w,16w+16).
// Each wave computes the i,f,g,o tiles itself -> pointwise update stays in regs.
// B tiles are double-buffered through LDS: one barrier per K-step, next tile's
// global load issued right after the barrier (hidden behind 4 WMMAs).
// ---------------------------------------------------------------------------
__global__ __launch_bounds__(128)
void lstm_step_kernel(int t,
                      const bf16* __restrict__ x,       // (B,T,D) bf16
                      bf16*       __restrict__ hs,      // (B,T,H) bf16
                      const bf16* __restrict__ Wcat,    // (4H, KC) bf16, [Wx | Wh]
                      const float* __restrict__ biasSum,// (4H) = bx + bh
                      float*      __restrict__ C,       // (B,H) f32 cell state
                      float*      __restrict__ hid,     // (B,T,H) f32 = tanh(h)
                      const bf16* __restrict__ zrow)    // 1024 zeros (t==0 h_prev)
{
  __shared__ __align__(64) bf16 Bs[2][BTILE];           // double-buffered, 8 KB

  const int lane = threadIdx.x & 31;
  const int wave = threadIdx.x >> 5;                    // 0..3 -> batch tile
  const int n0   = blockIdx.x * 16;                     // hidden-column tile

  v8f acc[4] = {};                                      // i, f, g, o accumulators

  // Branch-free A-operand addressing: kb<8 reads x_t, kb>=8 reads h_{t-1}
  // (pre-shifted by -DD); t==0 uses a zero-filled row instead of h.
  const int   bRow  = wave * 16 + (lane & 15);
  const bf16* xrow  = x + ((size_t)bRow * TT + t) * DD;
  const bf16* hbase = ((t > 0) ? (hs + ((size_t)bRow * TT + (t - 1)) * HH) : zrow) - DD;

  // Staging geometry: 128 threads copy 128 chunks of 16 bf16 (4 KB tile set).
  const int   e     = threadIdx.x * 16;
  const int   srow  = e >> 5;                           // 0..63 = gate*16 + r
  const int   soff  = e & 31;                           // 0 or 16
  const bf16* wsrc  = Wcat + (size_t)((srow >> 4) * HH + n0 + (srow & 15)) * KC + soff;
  bf16*       sdst  = &Bs[0][srow * 32 + soff];

  // Per-lane B-fragment base inside a staged tile set.
  const int bfoff = (lane & 15) * 32 + ((lane < 16) ? 0 : 16);

  v16bf gstage = *(const v16bf*)wsrc;                   // prologue: tile kb=0

  for (int kb = 0; kb < NKB; ++kb) {
    const int p = kb & 1;
    *(v16bf*)(sdst + p * BTILE) = gstage;               // publish tile kb
    __syncthreads();

    if (kb + 1 < NKB)                                   // fetch tile kb+1 (L2-resident)
      gstage = *(const v16bf*)(wsrc + (kb + 1) * 32);
    if (kb + 2 < NKB)
      __builtin_prefetch(wsrc + (kb + 2) * 32, 0, 3);   // near-scope prefetch

    const v16bf afrag = load_a_frag(((kb < 8) ? xrow : hbase) + kb * 32, lane);

    const bf16* bbase = &Bs[p][bfoff];
#pragma unroll
    for (int q = 0; q < 4; ++q) {
      const v16bf bfrag = *(const v16bf*)(bbase + q * (16 * 32));
      acc[q] = __builtin_amdgcn_wmma_f32_16x16x32_bf16(false, afrag, false, bfrag,
                                                       (short)0, acc[q], false, false);
    }
  }

  // ---- pointwise LSTM update on the WMMA C/D layout
  const int   n  = n0 + (lane & 15);
  const float bi = biasSum[0 * HH + n];
  const float bf = biasSum[1 * HH + n];
  const float bg = biasSum[2 * HH + n];
  const float bo = biasSum[3 * HH + n];
  const int   mbase = wave * 16 + ((lane < 16) ? 0 : 8);

#pragma unroll
  for (int r = 0; r < 8; ++r) {
    const int   m  = mbase + r;                         // batch index
    const float gi = sigmoidf_(acc[0][r] + bi);
    const float gf = sigmoidf_(acc[1][r] + bf);
    const float gg = sigmoidf_(acc[2][r] + bg);        // sigmoid on cell gate (per source)
    const float go = sigmoidf_(acc[3][r] + bo);

    float* cp = C + (size_t)m * HH + n;
    const float cnew = (*cp) * gf + gi * gg;
    *cp = cnew;

    const float h   = go * tanhf(cnew);
    const size_t ix = ((size_t)m * TT + t) * HH + n;
    hs[ix]  = (bf16)h;                                  // raw h (bf16) for next step & head
    hid[ix] = tanhf(h);                                 // hidden output = tanh(h)
  }
}

// ---------------------------------------------------------------------------
// Output head: out = hs * Wout^T + bout   (32768 x 256 x 1024, bf16 WMMA)
// 8 waves/block, one 16-row M tile per wave, 16-wide N tile per block.
// ---------------------------------------------------------------------------
__global__ __launch_bounds__(256)
void out_head_kernel(const bf16* __restrict__ hs,      // (B*T, H) bf16
                     const bf16* __restrict__ Wout,    // (D, H) bf16
                     const float* __restrict__ bout,   // (D)
                     float*      __restrict__ out)     // (B*T, D) f32
{
  const int lane  = threadIdx.x & 31;
  const int wave  = threadIdx.x >> 5;                  // 0..7
  const int ntile = blockIdx.x & 15;                   // D/16 tiles
  const int mblk  = blockIdx.x >> 4;
  const int n0    = ntile * 16;
  const int m0    = (mblk * 8 + wave) * 16;

  v8f acc = {};
  const bf16* arow = hs   + (size_t)(m0 + (lane & 15)) * HH;
  const bf16* wrow = Wout + (size_t)(n0 + (lane & 15)) * HH + ((lane < 16) ? 0 : 16);

  for (int kb = 0; kb < HH / 32; ++kb) {
    const v16bf afrag = load_a_frag(arow + kb * 32, lane);
    const v16bf bfrag = *(const v16bf*)(wrow + kb * 32);
    acc = __builtin_amdgcn_wmma_f32_16x16x32_bf16(false, afrag, false, bfrag,
                                                  (short)0, acc, false, false);
  }

  const int   n    = n0 + (lane & 15);
  const float bias = bout[n];
  const int   mb   = m0 + ((lane < 16) ? 0 : 8);
#pragma unroll
  for (int r = 0; r < 8; ++r)
    out[(size_t)(mb + r) * DD + n] = acc[r] + bias;
}

// ---------------------------------------------------------------------------
// Prep kernels (run every launch -> deterministic, no cached state)
// ---------------------------------------------------------------------------
__global__ void cvt_bf16_kernel(const float* __restrict__ s, bf16* __restrict__ d, int n) {
  int i = blockIdx.x * blockDim.x + threadIdx.x;
  const int stride = gridDim.x * blockDim.x;
  for (; i < n; i += stride) d[i] = (bf16)s[i];
}

__global__ void build_wcat_kernel(const float* __restrict__ Wx, const float* __restrict__ Wh,
                                  bf16* __restrict__ Wcat) {
  int i = blockIdx.x * blockDim.x + threadIdx.x;
  const int stride = gridDim.x * blockDim.x;
  const int n = G4 * KC;
  for (; i < n; i += stride) {
    const int j = i / KC, k = i % KC;
    const float v = (k < DD) ? Wx[j * DD + k] : Wh[j * HH + (k - DD)];
    Wcat[i] = (bf16)v;
  }
}

__global__ void bias_zero_kernel(const float* __restrict__ bx, const float* __restrict__ bh,
                                 float* __restrict__ biasSum, float* __restrict__ C,
                                 bf16* __restrict__ zrow) {
  const int i = blockIdx.x * blockDim.x + threadIdx.x;
  const int stride = gridDim.x * blockDim.x;
  for (int j = i; j < G4; j += stride) biasSum[j] = bx[j] + bh[j];
  for (int j = i; j < BB * HH; j += stride) C[j] = 0.0f;
  for (int j = i; j < HH; j += stride) zrow[j] = (bf16)0.0f;
}

// ---------------------------------------------------------------------------
extern "C" void kernel_launch(void* const* d_in, const int* in_sizes, int n_in,
                              void* d_out, int out_size, void* d_ws, size_t ws_size,
                              hipStream_t stream) {
  (void)in_sizes; (void)n_in; (void)out_size; (void)ws_size;

  const float* inputs = (const float*)d_in[0];
  const float* Wx     = (const float*)d_in[1];
  const float* bx     = (const float*)d_in[2];
  const float* Wh     = (const float*)d_in[3];
  const float* bh     = (const float*)d_in[4];
  const float* Wout   = (const float*)d_in[5];
  const float* bout   = (const float*)d_in[6];

  char* ws = (char*)d_ws;
  size_t off = 0;
  bf16*  x_bf    = (bf16*)(ws + off);  off += (size_t)BB * TT * DD * 2;  // 16.8 MB
  bf16*  Wcat    = (bf16*)(ws + off);  off += (size_t)G4 * KC * 2;       // 10.5 MB
  bf16*  Wout_bf = (bf16*)(ws + off);  off += (size_t)DD * HH * 2;       // 0.5 MB
  float* biasSum = (float*)(ws + off); off += (size_t)G4 * 4;
  float* C       = (float*)(ws + off); off += (size_t)BB * HH * 4;
  bf16*  zrow    = (bf16*)(ws + off);  off += (size_t)HH * 2;
  bf16*  hs      = (bf16*)(ws + off);  off += (size_t)BB * TT * HH * 2;  // 64 MB

  float* out = (float*)d_out;                       // (B,T,D)
  float* hid = out + (size_t)BB * TT * DD;          // (B,T,H)

  cvt_bf16_kernel<<<2048, 256, 0, stream>>>(inputs, x_bf, BB * TT * DD);
  cvt_bf16_kernel<<<1024, 256, 0, stream>>>(Wout, Wout_bf, DD * HH);
  build_wcat_kernel<<<2048, 256, 0, stream>>>(Wx, Wh, Wcat);
  bias_zero_kernel<<<256, 256, 0, stream>>>(bx, bh, biasSum, C, zrow);

  for (int t = 0; t < TT; ++t)
    lstm_step_kernel<<<HH / 16, 128, 0, stream>>>(t, x_bf, hs, Wcat, biasSum, C, hid, zrow);

  out_head_kernel<<<(BB * TT / (16 * 8)) * (DD / 16), 256, 0, stream>>>(hs, Wout_bf, bout, out);
}
